// SelfAttention_20057497272750
// MI455X (gfx1250) — matrix operand order repaired and checked by reading, not compiled, not run on hardware
//
#include <hip/hip_runtime.h>
#include <hip/hip_bf16.h>
#include <math.h>

typedef _Float16 half16 __attribute__((ext_vector_type(16)));
typedef _Float16 half8  __attribute__((ext_vector_type(8)));
typedef float    f32x8  __attribute__((ext_vector_type(8)));

#define WMMA16(a, b, c) \
    __builtin_amdgcn_wmma_f32_16x16x32_f16(false, (a), false, (b), (short)0, (c), false, false)

// Problem constants (match reference setup_inputs)
#define BB_   2
#define CC_   256
#define SS_   4096
#define NH_   4
#define HD_   64
#define G3_   768
#define GRP_  32
#define CPG_  8     // channels per group

#define KP_   72    // K-tile LDS row pitch in halves (64 + 8 pad)
#define VP_   40    // V-tile LDS row pitch in halves (32 + 8 pad)

// Load f16 A-matrix fragment: elements 0..7 from p[0..7], 8..15 from p[16..23].
// Caller passes p already offset by (lane>=16 ? 8 : 0) so the ISA A-layout
// (lanes 0-15: K 0-7 & 16-23; lanes 16-31: K 8-15 & 24-31) is produced.
__device__ __forceinline__ half16 load_a_f16(const _Float16* p) {
    half8 lo = *(const half8*)(p);
    half8 hi = *(const half8*)(p + 16);
    return __builtin_shufflevector(lo, hi, 0,1,2,3,4,5,6,7,8,9,10,11,12,13,14,15);
}

// CDNA5 async DMA: global memory -> LDS, 16 bytes, tracked by ASYNCcnt.
__device__ __forceinline__ void async_g2l_b128(uint32_t lds_off, const void* gptr) {
    asm volatile("global_load_async_to_lds_b128 %0, %1, off"
                 :: "v"(lds_off), "v"((uint64_t)(uintptr_t)gptr) : "memory");
}
__device__ __forceinline__ void wait_async0() {
    asm volatile("s_wait_asynccnt 0x0" ::: "memory");
}

// ---------------------------------------------------------------------------
// K0: fp32 -> f16 elementwise (weight pre-conversion, done once per launch).
// ---------------------------------------------------------------------------
__global__ __launch_bounds__(256) void k_cvt_f16(const float* __restrict__ src,
                                                 _Float16* __restrict__ dst, int n) {
    int i = blockIdx.x * 256 + threadIdx.x;
    if (i < n) dst[i] = (_Float16)src[i];
}

// ---------------------------------------------------------------------------
// K1: GroupNorm statistics. One block per (b, group).
// ---------------------------------------------------------------------------
__global__ __launch_bounds__(256) void k_gn_stats(const float* __restrict__ x,
                                                  float* __restrict__ stats) {
    const int bg = blockIdx.x;                       // b*GRP_ + g
    const float* base = x + (size_t)bg * CPG_ * SS_; // group channels contiguous
    const int N = CPG_ * SS_;
    float sum = 0.f, sq = 0.f;
    for (int i = threadIdx.x; i < N; i += 256) {
        float v = base[i];
        sum += v; sq += v * v;
    }
    __shared__ float ssum[256], ssq[256];
    ssum[threadIdx.x] = sum; ssq[threadIdx.x] = sq;
    __syncthreads();
    for (int off = 128; off > 0; off >>= 1) {
        if (threadIdx.x < off) {
            ssum[threadIdx.x] += ssum[threadIdx.x + off];
            ssq[threadIdx.x]  += ssq[threadIdx.x + off];
        }
        __syncthreads();
    }
    if (threadIdx.x == 0) {
        float mean = ssum[0] / (float)N;
        float var  = ssq[0] / (float)N - mean * mean;
        stats[bg * 2 + 0] = mean;
        stats[bg * 2 + 1] = rsqrtf(var + 1e-5f);
    }
}

// ---------------------------------------------------------------------------
// K2: apply GroupNorm + transpose to f16 normT[b][s][c], LDS-tiled.
// ---------------------------------------------------------------------------
__global__ __launch_bounds__(256) void k_gn_apply_t(const float* __restrict__ x,
                                                    const float* __restrict__ stats,
                                                    const float* __restrict__ w,
                                                    const float* __restrict__ bia,
                                                    _Float16* __restrict__ normT) {
    __shared__ _Float16 tile[16][257];               // padded: no bank conflicts
    const int blk = blockIdx.x;                      // B * 16 * 16
    const int b  = blk / 256;
    const int rr = blk % 256;
    const int c0 = (rr / 16) * 16;
    const int s0 = (rr % 16) * 256;
    const int t = threadIdx.x;
    const int s = s0 + t;
#pragma unroll
    for (int ci = 0; ci < 16; ++ci) {
        const int c = c0 + ci;
        const int bg = b * GRP_ + c / CPG_;
        const float mean = stats[bg * 2 + 0];
        const float rstd = stats[bg * 2 + 1];
        const float sc = rstd * w[c];
        const float sh = bia[c] - mean * sc;
        tile[ci][t] = (_Float16)(x[((size_t)b * CC_ + c) * SS_ + s] * sc + sh);
    }
    __syncthreads();
    half16 hv;
#pragma unroll
    for (int ci = 0; ci < 16; ++ci) hv[ci] = tile[ci][t];
    *(half16*)(normT + ((size_t)b * SS_ + s) * CC_ + c0) = hv;
}

// ---------------------------------------------------------------------------
// GEMM inner loop: 16(o) x 64(s) strip, K=256.
// ---------------------------------------------------------------------------
__device__ __forceinline__ void gemm_k256(const _Float16* __restrict__ arow,
                                          const _Float16* __restrict__ brow,
                                          f32x8 acc[4]) {
#pragma unroll
    for (int kc = 0; kc < 8; ++kc) {
        half16 A = load_a_f16(arow + kc * 32);
#pragma unroll
        for (int j = 0; j < 4; ++j) {
            half16 Bm = load_a_f16(brow + (size_t)j * 16 * CC_ + kc * 32);
            acc[j] = WMMA16(A, Bm, acc[j]);
        }
    }
}

// ---------------------------------------------------------------------------
// K3: QKV projection GEMM (per batch: M=768, N=4096, K=256) via WMMA f16.
// Q,K -> qkT[b][s][o] (transposed, b128 stores); V -> v[b][h][c][t].
// ---------------------------------------------------------------------------
__global__ __launch_bounds__(128) void k_qkv_gemm(const _Float16* __restrict__ Wh,
                                                  const _Float16* __restrict__ normT,
                                                  _Float16* __restrict__ qkT,
                                                  _Float16* __restrict__ vbuf) {
    const int wave = threadIdx.x >> 5, lane = threadIdx.x & 31;
    const int tid = blockIdx.x * 4 + wave;          // BB_ * 48 * 64 strips
    const int b  = tid / (48 * 64);
    const int rr = tid % (48 * 64);
    const int o0 = (rr / 64) * 16;
    const int s0 = (rr % 64) * 64;
    const int ls = lane & 15, sel = lane >> 4;

    const _Float16* arow = Wh + (size_t)(o0 + ls) * CC_ + sel * 8;
    const _Float16* brow = normT + ((size_t)b * SS_ + s0 + ls) * CC_ + sel * 8;

    f32x8 acc[4] = {};
    gemm_k256(arow, brow, acc);

    const int sec = (o0 % 192) / 64;                 // 0=q, 1=k, 2=v
    if (sec < 2) {
#pragma unroll
        for (int j = 0; j < 4; ++j) {
            _Float16* dst = qkT + ((size_t)b * SS_ + s0 + j * 16 + ls) * G3_ + o0 + sel * 8;
            half8 hv;
#pragma unroll
            for (int i = 0; i < 8; ++i) hv[i] = (_Float16)acc[j][i];
            *(half8*)dst = hv;
        }
    } else {
        const int h  = o0 / 192;
        const int c0 = (o0 % 192) - 128 + sel * 8;   // 0..63
        _Float16* vb = vbuf + (((size_t)b * NH_ + h) * HD_ + c0) * SS_ + s0 + ls;
#pragma unroll
        for (int j = 0; j < 4; ++j)
#pragma unroll
            for (int i = 0; i < 8; ++i) vb[(size_t)i * SS_ + j * 16] = (_Float16)acc[j][i];
    }
}

// ---------------------------------------------------------------------------
// K4: flash attention with async-to-LDS double buffering.
// Block = 4 waves on the same (b, h); each wave owns one 16-query tile.
// K (32x64) and V (64x32) tiles for each 32-key block are DMA'd to LDS once
// per block with global_load_async_to_lds_b128 (ASYNCcnt), one block ahead;
// fragments are then read from LDS (ds_load_b128, padded pitches).
// ---------------------------------------------------------------------------
__global__ __launch_bounds__(128) void k_flash(const _Float16* __restrict__ qkT,
                                               const _Float16* __restrict__ vbuf,
                                               _Float16* __restrict__ Ot) {
    __shared__ __align__(16) _Float16 kt[2][32 * KP_];
    __shared__ __align__(16) _Float16 vt[2][64 * VP_];

    const int wave = threadIdx.x >> 5, lane = threadIdx.x & 31;
    const int blk = blockIdx.x;                      // BB_ * NH_ * 64
    const int b  = blk / (NH_ * 64);
    const int rr = blk % (NH_ * 64);
    const int h  = rr / 64;
    const int s0 = (rr % 64) * 64 + wave * 16;
    const int ls = lane & 15, sel = lane >> 4;
    const bool lohalf = (sel == 0);
    const int t = threadIdx.x;                       // 0..127: staging lane id

    const _Float16* qrow  = qkT + ((size_t)b * SS_ + s0 + ls) * G3_ + h * 192;
    const _Float16* kbase = qkT + (size_t)b * SS_ * G3_ + h * 192 + 64;
    const _Float16* vb    = vbuf + ((size_t)b * NH_ + h) * (size_t)HD_ * SS_;

    // Stage one 32-key block (K tile 32x64, V tile 64x32) into LDS buffer `buf`.
    auto stage = [&](int buf, int t0) {
#pragma unroll
        for (int r = 0; r < 2; ++r) {                // K: 256 x 16B chunks
            const int chunk = t + 128 * r;
            const int row = chunk >> 3, col = (chunk & 7) * 8;   // halves
            async_g2l_b128((uint32_t)(uintptr_t)&kt[buf][row * KP_ + col],
                           kbase + (size_t)(t0 + row) * G3_ + col);
        }
#pragma unroll
        for (int r = 0; r < 2; ++r) {                // V: 256 x 16B chunks
            const int chunk = t + 128 * r;
            const int row = chunk >> 2, col = (chunk & 3) * 8;   // halves
            async_g2l_b128((uint32_t)(uintptr_t)&vt[buf][row * VP_ + col],
                           vb + (size_t)row * SS_ + t0 + col);
        }
    };

    // Loop-invariant Q B-fragments (K = channel chunks of 32)
    const half16 Bq0 = load_a_f16(qrow + sel * 8);
    const half16 Bq1 = load_a_f16(qrow + 32 + sel * 8);

    f32x8 acc0 = {}, acc1 = {}, acc2 = {}, acc3 = {};
    float mrun = -1e30f, lrun = 0.f;
    const float k2 = 0.0625f * 1.44269504088896f;    // (1/sqrt(C)) * log2(e)

    stage(0, 0);
    wait_async0();
    __syncthreads();

    for (int t0 = 0; t0 < SS_; t0 += 32) {
        const int cur = (t0 >> 5) & 1, nxt = cur ^ 1;
        // Prefetch the next key block into the other buffer (wrapped at end).
        stage(nxt, (t0 + 32) & (SS_ - 1));

        // K A-fragments from LDS (rows = key index, padded pitch).
        half16 Ak00 = load_a_f16(&kt[cur][(size_t)ls * KP_ + sel * 8]);
        half16 Ak01 = load_a_f16(&kt[cur][(size_t)ls * KP_ + 32 + sel * 8]);
        half16 Ak10 = load_a_f16(&kt[cur][(size_t)(16 + ls) * KP_ + sel * 8]);
        half16 Ak11 = load_a_f16(&kt[cur][(size_t)(16 + ls) * KP_ + 32 + sel * 8]);

        f32x8 T0 = {}, T1 = {};
        T0 = WMMA16(Ak00, Bq0, T0); T0 = WMMA16(Ak01, Bq1, T0);
        T1 = WMMA16(Ak10, Bq0, T1); T1 = WMMA16(Ak11, Bq1, T1);

        // V A-fragments from LDS (rows = channel, 32 key columns).
        half16 Av0 = load_a_f16(&vt[cur][(size_t)(ls)      * VP_ + sel * 8]);
        half16 Av1 = load_a_f16(&vt[cur][(size_t)(16 + ls) * VP_ + sel * 8]);
        half16 Av2 = load_a_f16(&vt[cur][(size_t)(32 + ls) * VP_ + sel * 8]);
        half16 Av3 = load_a_f16(&vt[cur][(size_t)(48 + ls) * VP_ + sel * 8]);

        // Online softmax (base-2): each lane owns 16 of the 32 t-values.
        float tmax = -1e30f;
#pragma unroll
        for (int i = 0; i < 8; ++i) {
            T0[i] *= k2; T1[i] *= k2;
            tmax = fmaxf(tmax, fmaxf(T0[i], T1[i]));
        }
        tmax = fmaxf(tmax, __shfl_xor(tmax, 16, 32));
        const float mnew = fmaxf(mrun, tmax);
        const float corr = __builtin_amdgcn_exp2f(mrun - mnew);
        float p0[8], p1[8], psum = 0.f;
#pragma unroll
        for (int i = 0; i < 8; ++i) {
            p0[i] = __builtin_amdgcn_exp2f(T0[i] - mnew);
            p1[i] = __builtin_amdgcn_exp2f(T1[i] - mnew);
            psum += p0[i] + p1[i];
        }
        psum += __shfl_xor(psum, 16, 32);
        lrun = lrun * corr + psum;
        mrun = mnew;
#pragma unroll
        for (int i = 0; i < 8; ++i) {
            acc0[i] *= corr; acc1[i] *= corr; acc2[i] *= corr; acc3[i] *= corr;
        }

        // Repack exp(P) from D-layout into the B-layout of the PV WMMA
        // (B[k=t][n=s]): table order, each lane keeps its own p0/p1.
        half16 Bp;
#pragma unroll
        for (int i = 0; i < 8; ++i) {
            Bp[i]     = (_Float16)p0[i];
            Bp[8 + i] = (_Float16)p1[i];
        }

        // PV: D[c][s] += V(M=c,K=t) x P^T(K=t,N=s)
        acc0 = WMMA16(Av0, Bp, acc0);
        acc1 = WMMA16(Av1, Bp, acc1);
        acc2 = WMMA16(Av2, Bp, acc2);
        acc3 = WMMA16(Av3, Bp, acc3);

        // Publish the prefetched buffer; also guarantees all waves are done
        // reading `cur` before it is overwritten next iteration.
        wait_async0();
        __syncthreads();
    }

    // Normalize and store transposed: Ot[b][s][h*64 + c]
    const float inv = 1.f / lrun;
    _Float16* orow = Ot + ((size_t)b * SS_ + s0 + ls) * CC_ + h * HD_ + sel * 8;
    f32x8 av[4] = {acc0, acc1, acc2, acc3};
#pragma unroll
    for (int ct = 0; ct < 4; ++ct) {
        half8 hv;
#pragma unroll
        for (int i = 0; i < 8; ++i) hv[i] = (_Float16)(av[ct][i] * inv);
        *(half8*)(orow + ct * 16) = hv;
    }
}

// ---------------------------------------------------------------------------
// K5: output projection (M=256, N=4096, K=256) + bias + residual, fp32 out.
// ---------------------------------------------------------------------------
__global__ __launch_bounds__(128) void k_out_proj(const _Float16* __restrict__ Owh,
                                                  const float* __restrict__ Ob,
                                                  const _Float16* __restrict__ Ot,
                                                  const float* __restrict__ x,
                                                  float* __restrict__ y) {
    const int wave = threadIdx.x >> 5, lane = threadIdx.x & 31;
    const int tid = blockIdx.x * 4 + wave;           // BB_ * 16 * 64 strips
    const int b  = tid / (16 * 64);
    const int rr = tid % (16 * 64);
    const int o0 = (rr / 64) * 16;
    const int s0 = (rr % 64) * 64;
    const int ls = lane & 15, sel = lane >> 4;

    const _Float16* arow = Owh + (size_t)(o0 + ls) * CC_ + sel * 8;
    const _Float16* brow = Ot + ((size_t)b * SS_ + s0 + ls) * CC_ + sel * 8;

    f32x8 acc[4] = {};
    gemm_k256(arow, brow, acc);

    const int o = o0 + sel * 8;                      // first of 8 rows this lane holds
#pragma unroll
    for (int j = 0; j < 4; ++j) {
        const size_t base = ((size_t)b * CC_ + o) * SS_ + s0 + j * 16 + ls;
#pragma unroll
        for (int i = 0; i < 8; ++i)
            y[base + (size_t)i * SS_] = acc[j][i] + Ob[o + i] + x[base + (size_t)i * SS_];
    }
}

// ---------------------------------------------------------------------------
extern "C" void kernel_launch(void* const* d_in, const int* in_sizes, int n_in,
                              void* d_out, int out_size, void* d_ws, size_t ws_size,
                              hipStream_t stream) {
    const float* x     = (const float*)d_in[0];
    const float* gn_w  = (const float*)d_in[1];
    const float* gn_b  = (const float*)d_in[2];
    const float* qkv_w = (const float*)d_in[3];
    const float* o_w   = (const float*)d_in[4];
    const float* o_b   = (const float*)d_in[5];
    float* out = (float*)d_out;

    char* ws = (char*)d_ws;
    float*    stats  = (float*)ws;                                   //   512 B
    _Float16* normT  = (_Float16*)(ws + 4096);                       //   4 MB
    _Float16* qkT    = normT + (size_t)BB_ * SS_ * CC_;              //  12 MB
    _Float16* vbuf   = qkT   + (size_t)BB_ * SS_ * G3_;              //   4 MB
    _Float16* Ot     = vbuf  + (size_t)BB_ * NH_ * HD_ * SS_;        //   4 MB
    _Float16* wqkv_h = Ot    + (size_t)BB_ * SS_ * CC_;              // 384 KB
    _Float16* ow_h   = wqkv_h + (size_t)G3_ * CC_;                   // 128 KB

    k_cvt_f16<<<(G3_ * CC_) / 256, 256, 0, stream>>>(qkv_w, wqkv_h, G3_ * CC_);
    k_cvt_f16<<<(CC_ * CC_) / 256, 256, 0, stream>>>(o_w, ow_h, CC_ * CC_);
    k_gn_stats<<<BB_ * GRP_, 256, 0, stream>>>(x, stats);
    k_gn_apply_t<<<BB_ * 16 * 16, 256, 0, stream>>>(x, stats, gn_w, gn_b, normT);
    k_qkv_gemm<<<(BB_ * 48 * 64) / 4, 128, 0, stream>>>(wqkv_h, normT, qkT, vbuf);
    k_flash<<<BB_ * NH_ * 64, 128, 0, stream>>>(qkT, vbuf, Ot);
    k_out_proj<<<(BB_ * 16 * 64) / 4, 128, 0, stream>>>(ow_h, o_b, Ot, x, out);
}
